// SpatialInteractionMix_15891378995280
// MI455X (gfx1250) — matrix operand order repaired and checked by reading, not compiled, not run on hardware
//
#include <hip/hip_runtime.h>
#include <hip/hip_bf16.h>

// ---------------------------------------------------------------------------
// SpatialInteractionMix for MI455X (gfx1250, wave32, WMMA 16x16x32 bf16)
//   - channel-mixing GEMMs on v_wmma_f32_16x16x32_bf16
//   - weights staged into LDS by the Tensor Data Mover (tensor_load_to_lds,
//     s_wait_tensorcnt) when they fit (Cin=256 -> 128KB <= 320KB WGP LDS)
//   - WKV recurrence as a 3-pass chunked stabilized exp-scan
// ---------------------------------------------------------------------------

#define Bc 4
#define Tc 16384
#define Cc 256
#define Hc 128
#define Wc 128
#define ROWS (Bc * Tc)      // 65536
#define SCH 64              // WKV chunks
#define LCH 256             // WKV chunk length

typedef __attribute__((ext_vector_type(16))) __bf16 v16bf;
typedef __attribute__((ext_vector_type(8)))  float  v8f;
typedef __attribute__((ext_vector_type(4)))  unsigned int u32x4;
typedef __attribute__((ext_vector_type(8)))  int i32x8;
typedef __attribute__((ext_vector_type(4)))  int i32x4;

union Frag16 { v16bf v; uint4 q[2]; };

// ------------------------------- conversions -------------------------------
__global__ __launch_bounds__(256)
void cvt2d(__bf16* __restrict__ dst, const float* __restrict__ src,
           int rows, int cols, int dstride, int doff) {
    size_t i = (size_t)blockIdx.x * 256 + threadIdx.x;
    size_t n = (size_t)rows * cols;
    if (i < n) {
        int r = (int)(i / cols);
        int c = (int)(i % cols);
        dst[(size_t)r * dstride + doff + c] = (__bf16)src[i];
    }
}

// ------------------------------- WMMA GEMM ---------------------------------
// out[row][n] = sum_k act[row][k] * W[n][k],  Cout fixed at 256.
// Block = 128 threads (4 waves); wave w covers N = w*64..w*64+63, block
// covers 32 rows (2 M-tiles per wave). USE_LDS: whole weight matrix is DMA'd
// into LDS by the TDM once per workgroup.
// MODE: 0 = bf16 out, 1 = f32 out, 2 = f32 out + residual, 3 = f32 sigmoid out
template <int MODE, int USE_LDS>
__global__ __launch_bounds__(128)
void gemm_wmma(const __bf16* __restrict__ act, const __bf16* __restrict__ Wt,
               float* __restrict__ outF, __bf16* __restrict__ outB,
               const float* __restrict__ res, int Cin) {
    extern __shared__ __bf16 ldsW[];

    if (USE_LDS) {
        if (threadIdx.x < 32) {
            // Tensor DMA descriptor (D#) per CDNA5 ISA ch.8:
            // group0: count=1 | lds_addr=0 | global_addr | type=2 ("image")
            unsigned long long ga = (unsigned long long)Wt;
            u32x4 g0 = { 1u,
                         0u,
                         (unsigned)(ga & 0xFFFFFFFFu),
                         (unsigned)((ga >> 32) & 0x01FFFFFFu) | (2u << 30) };
            // group1: data_size=1 (2B) | tensor_dim0=Cin | tensor_dim1=256 |
            //         tile_dim0=Cin | tile_dim1=256 | tensor_dim0_stride=Cin
            i32x8 g1 = { (int)0x00010000,
                         (int)((Cin & 0xFFFF) << 16),
                         (int)((Cc  & 0xFFFF) << 16),
                         (int)((Cin & 0xFFFF) << 16),
                         (int)(Cc & 0xFFFF),
                         (int)Cin,
                         0, 0 };
            i32x4 gz = {0, 0, 0, 0};
#if defined(__clang_major__) && (__clang_major__ >= 23)
            i32x8 gz8 = {0, 0, 0, 0, 0, 0, 0, 0};
            __builtin_amdgcn_tensor_load_to_lds(g0, g1, gz, gz, gz8, 0);
#else
            __builtin_amdgcn_tensor_load_to_lds(g0, g1, gz, gz, 0);
#endif
            __builtin_amdgcn_s_wait_tensorcnt((short)0);
        }
        __syncthreads();
    }

    const int mBase = blockIdx.x * 32;
    const int wave  = threadIdx.x >> 5;
    const int lane  = threadIdx.x & 31;
    const int l15   = lane & 15;
    const int g     = lane >> 4;
    const int nb    = wave * 64;

    const __bf16* aRow0 = act + (size_t)(mBase + l15) * Cin;
    const __bf16* aRow1 = act + (size_t)(mBase + 16 + l15) * Cin;

    v8f acc[2][4] = {};
    for (int kk = 0; kk < Cin; kk += 32) {
        Frag16 a0, a1;
        a0.q[0] = *(const uint4*)(aRow0 + kk + 16 * g);
        a0.q[1] = *(const uint4*)(aRow0 + kk + 16 * g + 8);
        a1.q[0] = *(const uint4*)(aRow1 + kk + 16 * g);
        a1.q[1] = *(const uint4*)(aRow1 + kk + 16 * g + 8);
#pragma unroll
        for (int j = 0; j < 4; ++j) {
            Frag16 bf;
            if (USE_LDS) {
                // ds_load_b128 from the TDM-staged copy
                bf.q[0] = *(const uint4*)(ldsW + (size_t)(nb + j * 16 + l15) * Cin + kk + 16 * g);
                bf.q[1] = *(const uint4*)(ldsW + (size_t)(nb + j * 16 + l15) * Cin + kk + 16 * g + 8);
            } else {
                const __bf16* wRow = Wt + (size_t)(nb + j * 16 + l15) * Cin + kk + 16 * g;
                bf.q[0] = *(const uint4*)(wRow);
                bf.q[1] = *(const uint4*)(wRow + 8);
            }
            acc[0][j] = __builtin_amdgcn_wmma_f32_16x16x32_bf16(
                false, a0.v, false, bf.v, (short)0, acc[0][j], false, false);
            acc[1][j] = __builtin_amdgcn_wmma_f32_16x16x32_bf16(
                false, a1.v, false, bf.v, (short)0, acc[1][j], false, false);
        }
    }
#pragma unroll
    for (int i = 0; i < 2; ++i) {
#pragma unroll
        for (int j = 0; j < 4; ++j) {
            int n = nb + j * 16 + l15;
#pragma unroll
            for (int r = 0; r < 8; ++r) {
                int m = mBase + i * 16 + 8 * g + r;
                size_t o = (size_t)m * Cc + n;
                float val = acc[i][j][r];
                if (MODE == 2) val += res[o];
                if (MODE == 3) val = 1.0f / (1.0f + __expf(-val));
                if (MODE == 0) outB[o] = (__bf16)val;
                else           outF[o] = val;
            }
        }
    }
}

// --------------------------- depthwise 3x3 convs ---------------------------
// mid: (B,T,C) bf16; dwcat: (B,T,3C) bf16. Dilations 1,2,3, zero 'same' pad.
__global__ __launch_bounds__(256)
void dw3x3(const __bf16* __restrict__ mid,
           const float* __restrict__ w1, const float* __restrict__ w2,
           const float* __restrict__ w3, __bf16* __restrict__ dwcat) {
    int t = blockIdx.x, b = blockIdx.y, c = threadIdx.x;
    int y0 = t >> 7, x0 = t & (Wc - 1);
    const __bf16* base = mid + ((size_t)b * Tc) * Cc + c;
    const float* wp[3] = {w1 + c * 9, w2 + c * 9, w3 + c * 9};
    size_t obase = ((size_t)(b * Tc + t)) * (3 * Cc) + c;
#pragma unroll
    for (int d = 1; d <= 3; ++d) {
        float acc = 0.0f;
#pragma unroll
        for (int i = 0; i < 3; ++i) {
#pragma unroll
            for (int j = 0; j < 3; ++j) {
                int yy = y0 + (i - 1) * d;
                int xx = x0 + (j - 1) * d;
                if (yy >= 0 && yy < Hc && xx >= 0 && xx < Wc)
                    acc += wp[d - 1][i * 3 + j] *
                           (float)base[((size_t)(yy * Wc + xx)) * Cc];
            }
        }
        dwcat[obase + (size_t)(d - 1) * Cc] = (__bf16)acc;
    }
}

// ------------------------------ channel mixing -----------------------------
__global__ __launch_bounds__(256)
void mix_kernel(const float* __restrict__ x, const float* __restrict__ xx,
                const float* __restrict__ mk, const float* __restrict__ mv,
                const float* __restrict__ mr, __bf16* __restrict__ xk,
                __bf16* __restrict__ xv, __bf16* __restrict__ xr) {
    size_t i = (size_t)blockIdx.x * 256 + threadIdx.x;
    if (i >= (size_t)ROWS * Cc) return;
    int c = (int)(i & (Cc - 1));
    float a = x[i], b = xx[i];
    float k = mk[c], v = mv[c], r = mr[c];
    xk[i] = (__bf16)(a * k + b * (1.0f - k));
    xv[i] = (__bf16)(a * v + b * (1.0f - v));
    xr[i] = (__bf16)(a * r + b * (1.0f - r));
}

// ------------------------------- WKV scan ----------------------------------
// State (aa,bb,pp) represents (A,B) = (aa,bb)*e^pp; per-step decay is additive
// in pp (pp += wneg), so a chunk of L steps decays incoming state by wneg*L.
__global__ __launch_bounds__(256)
void wkv_pass1(const float* __restrict__ k, const float* __restrict__ v,
               const float* __restrict__ decay, float* __restrict__ la,
               float* __restrict__ lb, float* __restrict__ lp) {
    int c = threadIdx.x, s = blockIdx.x, b = blockIdx.y;
    const float invT = 1.0f / (float)Tc;
    float wneg = -__expf(decay[c] * invT);
    float aa = 0.0f, bb = 0.0f, pp = -1e38f;
    size_t idx = ((size_t)b * Tc + (size_t)s * LCH) * Cc + c;
    for (int t = 0; t < LCH; ++t) {
        float kk = k[idx], vv = v[idx];
        idx += Cc;
        float ww = wneg + pp;
        float p  = fmaxf(ww, kk);
        float e1 = __expf(ww - p), e2 = __expf(kk - p);
        aa = e1 * aa + e2 * vv;
        bb = e1 * bb + e2;
        pp = p;
    }
    size_t o = ((size_t)b * SCH + s) * Cc + c;
    la[o] = aa; lb[o] = bb; lp[o] = pp;
}

__global__ __launch_bounds__(256)
void wkv_pass2(const float* __restrict__ decay, const float* __restrict__ la,
               const float* __restrict__ lb, const float* __restrict__ lp,
               float* __restrict__ ia, float* __restrict__ ib,
               float* __restrict__ ip) {
    int c = threadIdx.x, b = blockIdx.x;
    const float invT = 1.0f / (float)Tc;
    float wL = -__expf(decay[c] * invT) * (float)LCH;
    float a0 = 0.0f, b0 = 0.0f, p0 = -1e38f;
    for (int s = 0; s < SCH; ++s) {
        size_t o = ((size_t)b * SCH + s) * Cc + c;
        ia[o] = a0; ib[o] = b0; ip[o] = p0;
        float wwp = p0 + wL;
        float pl  = lp[o];
        float p   = fmaxf(wwp, pl);
        float e1  = __expf(wwp - p), e2 = __expf(pl - p);
        a0 = e1 * a0 + e2 * la[o];
        b0 = e1 * b0 + e2 * lb[o];
        p0 = p;
    }
}

__global__ __launch_bounds__(256)
void wkv_pass3(const float* __restrict__ k, const float* __restrict__ v,
               const float* __restrict__ decay, const float* __restrict__ first,
               const float* __restrict__ ia, const float* __restrict__ ib,
               const float* __restrict__ ip, float* __restrict__ y) {
    int c = threadIdx.x, s = blockIdx.x, b = blockIdx.y;
    const float invT = 1.0f / (float)Tc;
    float wneg = -__expf(decay[c] * invT);
    float u    = first[c] * invT;
    size_t o = ((size_t)b * SCH + s) * Cc + c;
    float aa = ia[o], bb = ib[o], pp = ip[o];
    size_t idx = ((size_t)b * Tc + (size_t)s * LCH) * Cc + c;
    for (int t = 0; t < LCH; ++t) {
        float kk = k[idx], vv = v[idx];
        float ww = u + kk;
        float p  = fmaxf(pp, ww);
        float e1 = __expf(pp - p), e2 = __expf(ww - p);
        y[idx] = (e1 * aa + e2 * vv) / (e1 * bb + e2);
        float ww2 = wneg + pp;
        float p2  = fmaxf(ww2, kk);
        float e1b = __expf(ww2 - p2), e2b = __expf(kk - p2);
        aa = e1b * aa + e2b * vv;
        bb = e1b * bb + e2b;
        pp = p2;
        idx += Cc;
    }
}

// --------------------------- LayerNorm + gate ------------------------------
__global__ __launch_bounds__(256)
void ln_gate(const float* __restrict__ y, const float* __restrict__ sr,
             const float* __restrict__ gamma, const float* __restrict__ beta,
             __bf16* __restrict__ gy) {
    __shared__ float sbuf[256];
    size_t row = blockIdx.x;
    int c = threadIdx.x;
    float v = y[row * Cc + c];
    sbuf[c] = v;
    __syncthreads();
    for (int s = 128; s > 0; s >>= 1) {
        if (c < s) sbuf[c] += sbuf[c + s];
        __syncthreads();
    }
    float mu = sbuf[0] * (1.0f / Cc);
    __syncthreads();
    float d = v - mu;
    sbuf[c] = d * d;
    __syncthreads();
    for (int s = 128; s > 0; s >>= 1) {
        if (c < s) sbuf[c] += sbuf[c + s];
        __syncthreads();
    }
    float var = sbuf[0] * (1.0f / Cc);
    float outv = d * rsqrtf(var + 1e-5f) * gamma[c] + beta[c];
    gy[row * Cc + c] = (__bf16)(outv * sr[row * Cc + c]);
}

// ------------------------------- host side ---------------------------------
extern "C" void kernel_launch(void* const* d_in, const int* in_sizes, int n_in,
                              void* d_out, int out_size, void* d_ws, size_t ws_size,
                              hipStream_t stream) {
    const float* x       = (const float*)d_in[0];
    const float* conv1_w = (const float*)d_in[1];
    const float* dw1_w   = (const float*)d_in[2];
    const float* dw2_w   = (const float*)d_in[3];
    const float* dw3_w   = (const float*)d_in[4];
    const float* proj1_w = (const float*)d_in[5];
    const float* proj2_w = (const float*)d_in[6];
    const float* proj3_w = (const float*)d_in[7];
    const float* decay   = (const float*)d_in[8];
    const float* first   = (const float*)d_in[9];
    const float* mix_k   = (const float*)d_in[10];
    const float* mix_v   = (const float*)d_in[11];
    const float* mix_r   = (const float*)d_in[12];
    const float* key_w   = (const float*)d_in[13];
    const float* value_w = (const float*)d_in[14];
    const float* recep_w = (const float*)d_in[15];
    const float* out_w   = (const float*)d_in[16];
    const float* gamma   = (const float*)d_in[17];
    const float* beta    = (const float*)d_in[18];
    float* out = (float*)d_out;

    const size_t MB = 1ull << 20;
    if (ws_size < 292 * MB) return;
    char* ws = (char*)d_ws;

    // weight region (0..4MB)
    __bf16* conv1_bf = (__bf16*)(ws + 0);
    __bf16* key_bf   = (__bf16*)(ws + 131072);
    __bf16* value_bf = (__bf16*)(ws + 262144);
    __bf16* recep_bf = (__bf16*)(ws + 393216);
    __bf16* outw_bf  = (__bf16*)(ws + 524288);
    __bf16* projcat  = (__bf16*)(ws + 655360);          // 256 x 768
    float*  la = (float*)(ws + 1048576);                // chunk states
    float*  lb = (float*)(ws + 1310720);
    float*  lp = (float*)(ws + 1572864);
    float*  ia = (float*)(ws + 1835008);
    float*  ib = (float*)(ws + 2097152);
    float*  ip = (float*)(ws + 2359296);

    // big slabs with staged reuse
    __bf16* x_bf  = (__bf16*)(ws + 4 * MB);             // 32MB; reused: xk
    __bf16* xk    = x_bf;
    __bf16* mid   = (__bf16*)(ws + 36 * MB);            // 32MB; reused: xv
    __bf16* xv    = mid;
    __bf16* dwcat = (__bf16*)(ws + 68 * MB);            // 96MB
    __bf16* xr    = dwcat;                              // first 32MB (after dwcat dead)
    __bf16* gy    = dwcat;                              // reused again after xr dead
    float*  kbuf  = (float*)(ws + 100 * MB);            // 64MB
    float*  xxbuf = (float*)(ws + 164 * MB);            // 64MB; reused: v
    float*  vbuf  = xxbuf;
    float*  srbuf = (float*)(ws + 228 * MB);            // 64MB
    float*  ybuf  = (float*)(ws + 4 * MB);              // 64MB over x_bf+mid (dead)

    const int n_act   = ROWS * Cc;
    const int gAct    = (n_act + 255) / 256;
    const int gW      = (Cc * Cc + 255) / 256;
    const int LDS_W   = Cc * Cc * 2;                    // 128KB TDM-staged weights

    // 1. conversions
    cvt2d<<<gW, 256, 0, stream>>>(conv1_bf, conv1_w, Cc, Cc, Cc, 0);
    cvt2d<<<gW, 256, 0, stream>>>(key_bf,   key_w,   Cc, Cc, Cc, 0);
    cvt2d<<<gW, 256, 0, stream>>>(value_bf, value_w, Cc, Cc, Cc, 0);
    cvt2d<<<gW, 256, 0, stream>>>(recep_bf, recep_w, Cc, Cc, Cc, 0);
    cvt2d<<<gW, 256, 0, stream>>>(outw_bf,  out_w,   Cc, Cc, Cc, 0);
    cvt2d<<<gW, 256, 0, stream>>>(projcat,  proj1_w, Cc, Cc, 3 * Cc, 0);
    cvt2d<<<gW, 256, 0, stream>>>(projcat,  proj2_w, Cc, Cc, 3 * Cc, Cc);
    cvt2d<<<gW, 256, 0, stream>>>(projcat,  proj3_w, Cc, Cc, 3 * Cc, 2 * Cc);
    cvt2d<<<gAct, 256, 0, stream>>>(x_bf, x, ROWS, Cc, Cc, 0);

    // 2. mid = conv1(x)                       [TDM weights -> LDS]
    gemm_wmma<0, 1><<<ROWS / 32, 128, LDS_W, stream>>>(x_bf, conv1_bf, nullptr, mid, nullptr, Cc);

    // 3. depthwise convs -> dwcat
    dw3x3<<<dim3(Tc, Bc), 256, 0, stream>>>(mid, dw1_w, dw2_w, dw3_w, dwcat);

    // 4. xx = projcat(dwcat) + x              [Cin=768: weights too big for LDS]
    gemm_wmma<2, 0><<<ROWS / 32, 128, 0, stream>>>(dwcat, projcat, xxbuf, nullptr, x, 3 * Cc);

    // 5. channel mixing
    mix_kernel<<<gAct, 256, 0, stream>>>(x, xxbuf, mix_k, mix_v, mix_r, xk, xv, xr);

    // 6. k, v, sr                             [TDM weights -> LDS]
    gemm_wmma<1, 1><<<ROWS / 32, 128, LDS_W, stream>>>(xk, key_bf,   kbuf,  nullptr, nullptr, Cc);
    gemm_wmma<1, 1><<<ROWS / 32, 128, LDS_W, stream>>>(xv, value_bf, vbuf,  nullptr, nullptr, Cc);
    gemm_wmma<3, 1><<<ROWS / 32, 128, LDS_W, stream>>>(xr, recep_bf, srbuf, nullptr, nullptr, Cc);

    // 7. chunked WKV scan
    wkv_pass1<<<dim3(SCH, Bc), 256, 0, stream>>>(kbuf, vbuf, decay, la, lb, lp);
    wkv_pass2<<<Bc, 256, 0, stream>>>(decay, la, lb, lp, ia, ib, ip);
    wkv_pass3<<<dim3(SCH, Bc), 256, 0, stream>>>(kbuf, vbuf, decay, first, ia, ib, ip, ybuf);

    // 8. LayerNorm + gate
    ln_gate<<<ROWS, 256, 0, stream>>>(ybuf, srbuf, gamma, beta, gy);

    // 9. output projection                    [TDM weights -> LDS]
    gemm_wmma<1, 1><<<ROWS / 32, 128, LDS_W, stream>>>(gy, outw_bf, out, nullptr, nullptr, Cc);
}
